// SwinTransformerBlock_16234976379245
// MI455X (gfx1250) — matrix operand order repaired and checked by reading, not compiled, not run on hardware
//
#include <hip/hip_runtime.h>
#include <hip/hip_bf16.h>

typedef __attribute__((ext_vector_type(16))) _Float16 v16h;
typedef __attribute__((ext_vector_type(8)))  float    v8f;

#define B_     32
#define H_     56
#define W_     56
#define C_     128
#define NH_    4
#define WS_    7
#define SHIFT_ 3
#define N_     49          // tokens per window
#define HD_    32          // head dim
#define HID_   512
#define NWIN_  64          // windows per image (8x8)
#define BW_    (B_ * NWIN_)   // 2048 windows total
#define T_     (BW_ * N_)     // 100352 tokens
#define SCALE_ 0.1767766952966369f
#define EPS_   1e-5f

// ---------------------------------------------------------------------------
// WMMA fragment helpers (CDNA5 wave32; A/B use contiguous k = 16*(lane/16)+s, C/D per CDNA5 ISA 7.12.2)
// ---------------------------------------------------------------------------
__device__ __forceinline__ v8f wmma_f32(v16h a, v16h b, v8f c) {
  return __builtin_amdgcn_wmma_f32_16x16x32_f16(false, a, false, b, (short)0, c,
                                                false, false);
}

// A: 16x32 f16 (MxK), row-major source, ld in halfs.
// lanes 0-15: M=lane, v0-7 -> K=0..15
// lanes16-31: M=lane-16, v0-7 -> K=16..31
__device__ __forceinline__ v16h load_A(const _Float16* src, int ld) {
  int lane = threadIdx.x & 31;
  int row  = lane & 15;
  int kh   = (lane >> 4) & 1;
  const _Float16* p = src + row * ld;
  v16h a;
#pragma unroll
  for (int v = 0; v < 8; ++v) {
    int kbase = kh * 16 + 2 * v;
    a[2 * v]     = p[kbase];
    a[2 * v + 1] = p[kbase + 1];
  }
  return a;
}

// B: 32x16 f16 (KxN) built from row-major W[Nout][K] (i.e. B = W^T tile).
// lane n (n<16): column n, v holds K=2v,2v+1 ; lane n+16: K=16+2v,16+2v+1
__device__ __forceinline__ v16h load_Bt(const _Float16* Wt, int ldw) {
  int lane = threadIdx.x & 31;
  int col  = lane & 15;
  int kh   = (lane >> 4) & 1;
  const _Float16* p = Wt + col * ldw + kh * 16;
  v16h b;
#pragma unroll
  for (int v = 0; v < 8; ++v) {
    b[2 * v]     = p[2 * v];
    b[2 * v + 1] = p[2 * v + 1];
  }
  return b;
}

// B: 32x16 f16 (KxN) from row-major V[K][N] directly.
__device__ __forceinline__ v16h load_Bd(const _Float16* V, int ldv) {
  int lane = threadIdx.x & 31;
  int col  = lane & 15;
  int kh   = (lane >> 4) & 1;
  v16h b;
#pragma unroll
  for (int v = 0; v < 8; ++v) {
    int k = kh * 16 + 2 * v;
    b[2 * v]     = V[k * ldv + col];
    b[2 * v + 1] = V[(k + 1) * ldv + col];
  }
  return b;
}

// C/D f32 16x16: lane&15 = N, (lane>>4)*8 + vgpr = M
__device__ __forceinline__ int c_ncol() { return threadIdx.x & 15; }
__device__ __forceinline__ int c_mbase() { return ((threadIdx.x >> 4) & 1) * 8; }

// ---------------------------------------------------------------------------
// K0: weight f32->f16 conversion + relative-position bias table gather
// ---------------------------------------------------------------------------
__global__ void k_prep(const float* qkv_w, const float* proj_w, const float* fc1_w,
                       const float* fc2_w, const float* rel_table, const int* rel_index,
                       _Float16* qkv_h, _Float16* proj_h, _Float16* fc1_h,
                       _Float16* fc2_h, float* bias) {
  int i = blockIdx.x * blockDim.x + threadIdx.x;
  int stride = gridDim.x * blockDim.x;
  for (int j = i; j < 384 * 128; j += stride) qkv_h[j] = (_Float16)qkv_w[j];
  for (int j = i; j < 128 * 128; j += stride) proj_h[j] = (_Float16)proj_w[j];
  for (int j = i; j < 512 * 128; j += stride) fc1_h[j] = (_Float16)fc1_w[j];
  for (int j = i; j < 128 * 512; j += stride) fc2_h[j] = (_Float16)fc2_w[j];
  for (int j = i; j < N_ * N_; j += stride) {
    int ri = rel_index[j];
#pragma unroll
    for (int h = 0; h < NH_; ++h)
      bias[h * N_ * N_ + j] = rel_table[ri * NH_ + h];
  }
}

// ---------------------------------------------------------------------------
// K1: LN1 + cyclic shift + window partition  -> xw f16 [T_, C_] (window order)
// one wave per token
// ---------------------------------------------------------------------------
__global__ void k_ln1(const float* __restrict__ x, const float* nw, const float* nb,
                      _Float16* __restrict__ xw) {
  int wave = threadIdx.x >> 5;
  int lane = threadIdx.x & 31;
  int t = blockIdx.x * 8 + wave;
  int wg = t / N_, n = t % N_;
  int b = wg / NWIN_, wi = wg % NWIN_;
  int wh = wi >> 3, ww = wi & 7;
  int r = n / WS_, c = n % WS_;
  int hs = wh * WS_ + r + SHIFT_;  if (hs >= H_) hs -= H_;
  int vs = ww * WS_ + c + SHIFT_;  if (vs >= W_) vs -= W_;
  const float* xp = x + ((size_t)b * H_ * W_ + hs * W_ + vs) * C_;
  float v0[4], s = 0.f, sq = 0.f;
#pragma unroll
  for (int i = 0; i < 4; ++i) {
    v0[i] = xp[lane * 4 + i];
    s += v0[i]; sq += v0[i] * v0[i];
  }
#pragma unroll
  for (int off = 16; off > 0; off >>= 1) {
    s += __shfl_xor(s, off); sq += __shfl_xor(sq, off);
  }
  float m  = s * (1.f / C_);
  float var = sq * (1.f / C_) - m * m;
  float rs = rsqrtf(var + EPS_);
  _Float16* op = xw + (size_t)t * C_;
#pragma unroll
  for (int i = 0; i < 4; ++i) {
    int ci = lane * 4 + i;
    op[ci] = (_Float16)((v0[i] - m) * rs * nw[ci] + nb[ci]);
  }
}

// ---------------------------------------------------------------------------
// K2: QKV GEMM (M=T_, K=128, N=384) + scatter to padded q/k/v [BW][NH][64][32]
// one wave per 16x64 output strip: A fragment reused across 4 WMMAs per K-step
// ---------------------------------------------------------------------------
__global__ void k_qkv(const _Float16* __restrict__ xw, const _Float16* __restrict__ wq,
                      const float* __restrict__ qkvb, _Float16* qb, _Float16* kb,
                      _Float16* vb) {
  int wave = threadIdx.x >> 5;
  int wid = blockIdx.x * 8 + wave;          // (T_/16) * 6 strip jobs
  int mt = wid / 6, ng = wid % 6;           // ng = group of 4 N-tiles (64 cols)
  const _Float16* Ab = xw + (size_t)mt * 16 * C_;
  __builtin_prefetch(wq + (size_t)ng * 64 * C_, 0, 0);
  v8f acc[4] = {};
#pragma unroll
  for (int k0 = 0; k0 < C_; k0 += 32) {
    v16h a = load_A(Ab + k0, C_);
#pragma unroll
    for (int j = 0; j < 4; ++j) {
      v16h b = load_Bt(wq + (size_t)(ng * 4 + j) * 16 * C_ + k0, C_);
      acc[j] = wmma_f32(a, b, acc[j]);
    }
  }
  int ncol = c_ncol(), mbase = c_mbase();
#pragma unroll
  for (int j = 0; j < 4; ++j) {
    int o = (ng * 4 + j) * 16 + ncol;
    int part = o >> 7;            // 0=q 1=k 2=v
    int head = (o >> 5) & 3;
    int hd = o & 31;
    float bias = qkvb[o];
    _Float16* dst = (part == 0) ? qb : (part == 1) ? kb : vb;
#pragma unroll
    for (int v = 0; v < 8; ++v) {
      int tok = mt * 16 + mbase + v;
      int wg = tok / N_, n = tok % N_;
      dst[(((size_t)wg * NH_ + head) * 64 + n) * HD_ + hd] =
          (_Float16)(acc[j][v] + bias);
    }
  }
}

// ---------------------------------------------------------------------------
// K3: windowed attention. block = 4 waves (one head each), one window/block.
// S = q k^T (16 WMMA), +scale +bias, mask cols>=49, softmax in LDS, P@V (16 WMMA)
// ---------------------------------------------------------------------------
__global__ void k_attn(const _Float16* __restrict__ qb, const _Float16* __restrict__ kb,
                       const _Float16* __restrict__ vb, const float* __restrict__ bias,
                       _Float16* __restrict__ attn_out) {
  extern __shared__ char smem[];
  float*    S = (float*)smem;                         // [NH][64][66] f32
  _Float16* P = (_Float16*)(smem + NH_ * 64 * 66 * 4);// [NH][64][66] f16
  int h = threadIdx.x >> 5;
  int lane = threadIdx.x & 31;
  int w = blockIdx.x;
  const _Float16* q = qb + ((size_t)w * NH_ + h) * 64 * HD_;
  const _Float16* k = kb + ((size_t)w * NH_ + h) * 64 * HD_;
  const _Float16* v = vb + ((size_t)w * NH_ + h) * 64 * HD_;
  float*    Sh = S + h * 64 * 66;
  _Float16* Ph = P + h * 64 * 66;
  const float* bh = bias + h * N_ * N_;
  int ncol = lane & 15, mbase = ((lane >> 4) & 1) * 8;

  // scores: A (q tile) reused across the 4 k^T tiles
#pragma unroll
  for (int mi = 0; mi < 4; ++mi) {
    v16h a = load_A(q + mi * 16 * HD_, HD_);
#pragma unroll
    for (int ni = 0; ni < 4; ++ni) {
      v16h b = load_Bt(k + ni * 16 * HD_, HD_);   // B = K^T tile
      v8f acc = {};
      acc = wmma_f32(a, b, acc);
#pragma unroll
      for (int vv = 0; vv < 8; ++vv) {
        int m = mi * 16 + mbase + vv;
        int n = ni * 16 + ncol;
        float val = (m < N_ && n < N_) ? acc[vv] * SCALE_ + bh[m * N_ + n] : -1e30f;
        Sh[m * 66 + n] = val;
      }
    }
  }
  // softmax (wave-private slice of LDS; DS ops in-order within wave)
  for (int r = lane; r < 64; r += 32) {
    float* row = Sh + r * 66;
    float mx = -1e30f;
    for (int c2 = 0; c2 < N_; ++c2) mx = fmaxf(mx, row[c2]);
    float sum = 0.f;
    for (int c2 = 0; c2 < N_; ++c2) { float e = __expf(row[c2] - mx); row[c2] = e; sum += e; }
    float inv = 1.f / sum;
    _Float16* prow = Ph + r * 66;
    for (int c2 = 0; c2 < 64; ++c2)
      prow[c2] = (c2 < N_) ? (_Float16)(row[c2] * inv) : (_Float16)0.f;
  }
  // out = P @ V   (M=64pad, K=64pad, N=32)
#pragma unroll
  for (int mi = 0; mi < 4; ++mi) {
    v16h a0 = load_A(Ph + mi * 16 * 66, 66);
    v16h a1 = load_A(Ph + mi * 16 * 66 + 32, 66);
#pragma unroll
    for (int nj = 0; nj < 2; ++nj) {
      v8f acc = {};
      v16h b0 = load_Bd(v + nj * 16, HD_);
      acc = wmma_f32(a0, b0, acc);
      v16h b1 = load_Bd(v + 32 * HD_ + nj * 16, HD_);
      acc = wmma_f32(a1, b1, acc);
#pragma unroll
      for (int vv = 0; vv < 8; ++vv) {
        int m = mi * 16 + mbase + vv;
        if (m < N_)
          attn_out[((size_t)w * N_ + m) * C_ + h * HD_ + nj * 16 + ncol] =
              (_Float16)acc[vv];
      }
    }
  }
}

// ---------------------------------------------------------------------------
// K4: proj GEMM (M=T_, K=128, N=128) + window reverse + un-shift + shortcut
// one wave per 16x64 output strip
// ---------------------------------------------------------------------------
__global__ void k_proj(const _Float16* __restrict__ ao, const _Float16* __restrict__ pw,
                       const float* __restrict__ pb, const float* __restrict__ x,
                       float* __restrict__ y) {
  int wave = threadIdx.x >> 5;
  int wid = blockIdx.x * 8 + wave;          // (T_/16) * 2 strip jobs
  int mt = wid >> 1, ng = wid & 1;
  const _Float16* Ab = ao + (size_t)mt * 16 * C_;
  __builtin_prefetch(pw + (size_t)ng * 64 * C_, 0, 0);
  v8f acc[4] = {};
#pragma unroll
  for (int k0 = 0; k0 < C_; k0 += 32) {
    v16h a = load_A(Ab + k0, C_);
#pragma unroll
    for (int j = 0; j < 4; ++j) {
      v16h b = load_Bt(pw + (size_t)(ng * 4 + j) * 16 * C_ + k0, C_);
      acc[j] = wmma_f32(a, b, acc[j]);
    }
  }
  int ncol = c_ncol(), mbase = c_mbase();
#pragma unroll
  for (int vv = 0; vv < 8; ++vv) {
    int tok = mt * 16 + mbase + vv;
    int wg = tok / N_, n = tok % N_;
    int b = wg / NWIN_, wi = wg % NWIN_;
    int wh = wi >> 3, ww = wi & 7;
    int r = n / WS_, c2 = n % WS_;
    int hi = wh * WS_ + r + SHIFT_;  if (hi >= H_) hi -= H_;
    int wj = ww * WS_ + c2 + SHIFT_; if (wj >= W_) wj -= W_;
    size_t row = (size_t)b * H_ * W_ + hi * W_ + wj;
#pragma unroll
    for (int j = 0; j < 4; ++j) {
      int o = (ng * 4 + j) * 16 + ncol;
      y[row * C_ + o] = x[row * C_ + o] + acc[j][vv] + pb[o];
    }
  }
}

// ---------------------------------------------------------------------------
// K5: fused MLP. block = 8 waves over 64 tokens.
// LN2 -> A1(LDS f16) -> fc1 WMMA -> exact GELU -> Hmid(LDS f16) -> fc2 WMMA -> +y
// LDS: A1 [64][136] f16 (17 KB) + Hmid [64][520] f16 (65 KB)
// Each GEMM job = 16x64 strip (A fragment reused across 4 WMMAs per K-step)
// ---------------------------------------------------------------------------
__global__ void k_mlp(const float* __restrict__ y, const float* n2w, const float* n2b,
                      const _Float16* __restrict__ f1w, const float* __restrict__ f1b,
                      const _Float16* __restrict__ f2w, const float* __restrict__ f2b,
                      float* __restrict__ out) {
  extern __shared__ char smem[];
  _Float16* A1 = (_Float16*)smem;                 // [64][136]
  _Float16* Hm = (_Float16*)(smem + 64 * 136 * 2);// [64][520]
  int wave = threadIdx.x >> 5, lane = threadIdx.x & 31;
  size_t tok0 = (size_t)blockIdx.x * 64;
  // LN2 -> LDS f16
  for (int rr = 0; rr < 8; ++rr) {
    int r = wave * 8 + rr;
    const float* yp = y + (tok0 + r) * C_;
    float v0[4], s = 0.f, sq = 0.f;
#pragma unroll
    for (int i = 0; i < 4; ++i) {
      v0[i] = yp[lane * 4 + i];
      s += v0[i]; sq += v0[i] * v0[i];
    }
#pragma unroll
    for (int off = 16; off > 0; off >>= 1) {
      s += __shfl_xor(s, off); sq += __shfl_xor(sq, off);
    }
    float m = s * (1.f / C_), var = sq * (1.f / C_) - m * m;
    float rs = rsqrtf(var + EPS_);
#pragma unroll
    for (int i = 0; i < 4; ++i) {
      int ci = lane * 4 + i;
      A1[r * 136 + ci] = (_Float16)((v0[i] - m) * rs * n2w[ci] + n2b[ci]);
    }
  }
  __syncthreads();
  int ncol = c_ncol(), mbase = c_mbase();
  // fc1 + exact GELU: 32 strip jobs (4 m-tiles x 8 groups of 64 cols)
  for (int job = wave; job < 32; job += 8) {
    int mi = job >> 3, ng = job & 7;
    v8f acc[4] = {};
#pragma unroll
    for (int k0 = 0; k0 < C_; k0 += 32) {
      v16h a = load_A(A1 + mi * 16 * 136 + k0, 136);
#pragma unroll
      for (int j = 0; j < 4; ++j) {
        v16h b = load_Bt(f1w + (size_t)(ng * 4 + j) * 16 * C_ + k0, C_);
        acc[j] = wmma_f32(a, b, acc[j]);
      }
    }
#pragma unroll
    for (int j = 0; j < 4; ++j) {
#pragma unroll
      for (int vv = 0; vv < 8; ++vv) {
        float xg = acc[j][vv] + f1b[(ng * 4 + j) * 16 + ncol];
        float g = 0.5f * xg * (1.f + erff(xg * 0.70710678118f));
        Hm[(mi * 16 + mbase + vv) * 520 + (ng * 4 + j) * 16 + ncol] = (_Float16)g;
      }
    }
  }
  __syncthreads();
  // fc2 + residual: 8 strip jobs (4 m-tiles x 2 groups), one per wave
  {
    int mi = wave >> 1, ng = wave & 1;
    v8f acc[4] = {};
#pragma unroll
    for (int k0 = 0; k0 < HID_; k0 += 32) {
      v16h a = load_A(Hm + mi * 16 * 520 + k0, 520);
#pragma unroll
      for (int j = 0; j < 4; ++j) {
        v16h b = load_Bt(f2w + (size_t)(ng * 4 + j) * 16 * HID_ + k0, HID_);
        acc[j] = wmma_f32(a, b, acc[j]);
      }
    }
#pragma unroll
    for (int vv = 0; vv < 8; ++vv) {
      size_t row = tok0 + mi * 16 + mbase + vv;
#pragma unroll
      for (int j = 0; j < 4; ++j) {
        int o = (ng * 4 + j) * 16 + ncol;
        out[row * C_ + o] = y[row * C_ + o] + acc[j][vv] + f2b[o];
      }
    }
  }
}

// ---------------------------------------------------------------------------
extern "C" void kernel_launch(void* const* d_in, const int* in_sizes, int n_in,
                              void* d_out, int out_size, void* d_ws, size_t ws_size,
                              hipStream_t stream) {
  const float* x        = (const float*)d_in[0];
  // d_in[1] = attn_mask (unused by reference)
  const float* norm1_w  = (const float*)d_in[2];
  const float* norm1_b  = (const float*)d_in[3];
  const float* qkv_w    = (const float*)d_in[4];
  const float* qkv_b    = (const float*)d_in[5];
  const float* rel_tab  = (const float*)d_in[6];
  const int*   rel_idx  = (const int*)d_in[7];
  const float* proj_w   = (const float*)d_in[8];
  const float* proj_b   = (const float*)d_in[9];
  const float* norm2_w  = (const float*)d_in[10];
  const float* norm2_b  = (const float*)d_in[11];
  const float* fc1_w    = (const float*)d_in[12];
  const float* fc1_b    = (const float*)d_in[13];
  const float* fc2_w    = (const float*)d_in[14];
  const float* fc2_b    = (const float*)d_in[15];
  float* out = (float*)d_out;

  char* ws = (char*)d_ws;
  _Float16* qkv_h  = (_Float16*)(ws);                  //  98304 B
  _Float16* proj_h = (_Float16*)(ws + 98304);          //  32768 B
  _Float16* f1_h   = (_Float16*)(ws + 131072);         // 131072 B
  _Float16* f2_h   = (_Float16*)(ws + 262144);         // 131072 B
  float*    biasb  = (float*)   (ws + 393216);         //  38416 B
  _Float16* xw     = (_Float16*)(ws + 431872);         // 25690112 B
  _Float16* qbuf   = (_Float16*)(ws + 26121984);       // 33554432 B
  _Float16* kbuf   = (_Float16*)(ws + 59676416);       // 33554432 B
  _Float16* vbuf   = (_Float16*)(ws + 93230848);       // 33554432 B
  _Float16* attn_o = xw;  // xw dead after k_qkv -> reuse for attention output

  k_prep<<<256, 256, 0, stream>>>(qkv_w, proj_w, fc1_w, fc2_w, rel_tab, rel_idx,
                                  qkv_h, proj_h, f1_h, f2_h, biasb);

  k_ln1<<<T_ / 8, 256, 0, stream>>>(x, norm1_w, norm1_b, xw);

  // (T_/16)*6 strip jobs / 8 waves per block = 4704 blocks
  k_qkv<<<(T_ / 16) * 6 / 8, 256, 0, stream>>>(xw, qkv_h, qkv_b, qbuf, kbuf, vbuf);

  const int attn_lds = NH_ * 64 * 66 * 4 + NH_ * 64 * 66 * 2;  // 101376 B
  k_attn<<<BW_, 128, attn_lds, stream>>>(qbuf, kbuf, vbuf, biasb, attn_o);

  // (T_/16)*2 strip jobs / 8 waves per block = 1568 blocks
  k_proj<<<(T_ / 16) * 2 / 8, 256, 0, stream>>>(attn_o, proj_h, proj_b, x, out);

  const int mlp_lds = 64 * 136 * 2 + 64 * 520 * 2;  // 83968 B
  k_mlp<<<T_ / 64, 256, mlp_lds, stream>>>(out, norm2_w, norm2_b, f1_h, fc1_b,
                                           f2_h, fc2_b, out);
}
